// EncoderLayer_5446018531529
// MI455X (gfx1250) — matrix operand order repaired and checked by reading, not compiled, not run on hardware
//
#include <hip/hip_runtime.h>

// ---------------------------------------------------------------------------
// Types for CDNA5 WMMA (wave32): v_wmma_f32_16x16x32_bf16
// ---------------------------------------------------------------------------
typedef __attribute__((ext_vector_type(16))) __bf16 v16bf;
typedef __attribute__((ext_vector_type(8)))  __bf16 v8bf;
typedef __attribute__((ext_vector_type(8)))  float  v8f;
typedef __attribute__((ext_vector_type(4)))  int    v4i;

#define B_  4
#define S_  2048
#define D_  1024
#define H_  16
#define HD_ 64
#define PF_ 4096
#define M_  (B_ * S_)   // 8192 rows

__device__ __forceinline__ v8f vzero8() {
  v8f z;
#pragma unroll
  for (int i = 0; i < 8; ++i) z[i] = 0.0f;
  return z;
}

__device__ __forceinline__ v16bf ld16bf(const __bf16* p) {
  v8bf lo = *(const v8bf*)(p);
  v8bf hi = *(const v8bf*)(p + 8);
  return __builtin_shufflevector(lo, hi, 0, 1, 2, 3, 4, 5, 6, 7,
                                 8, 9, 10, 11, 12, 13, 14, 15);
}

__device__ __forceinline__ v16bf ld8x2bf(const __bf16* p0, const __bf16* p1) {
  v8bf lo = *(const v8bf*)(p0);
  v8bf hi = *(const v8bf*)(p1);
  return __builtin_shufflevector(lo, hi, 0, 1, 2, 3, 4, 5, 6, 7,
                                 8, 9, 10, 11, 12, 13, 14, 15);
}

__device__ __forceinline__ v8f wmma_bf16(v16bf a, v16bf b, v8f c) {
  // (neg_a, A, neg_b, B, c_mod, C, reuse_a, reuse_b)
  return __builtin_amdgcn_wmma_f32_16x16x32_bf16(false, a, false, b,
                                                 (short)0, c, false, false);
}

// ---------------------------------------------------------------------------
// Async memory->LDS copy (CDNA5 GLOBAL_LOAD_ASYNC_TO_LDS_B128, ASYNCcnt).
// Builtin signature (from hipcc diagnostic): (as1 v4i*, as3 v4i*, imm, imm).
// ---------------------------------------------------------------------------
#if defined(__gfx1250__) && __has_builtin(__builtin_amdgcn_global_load_async_to_lds_b128)
#define ASYNC_MODE 2
#elif defined(__gfx1250__)
#define ASYNC_MODE 1
#else
#define ASYNC_MODE 0
#endif

__device__ __forceinline__ void copy16_async(const __bf16* g, __bf16* l) {
#if ASYNC_MODE == 2
  __builtin_amdgcn_global_load_async_to_lds_b128(
      (__attribute__((address_space(1))) v4i*)g,
      (__attribute__((address_space(3))) v4i*)l, 0, 0);
#elif ASYNC_MODE == 1
  unsigned int loff =
      (unsigned int)(unsigned long)(__attribute__((address_space(3))) __bf16*)l;
  unsigned long long ga = (unsigned long long)g;
  asm volatile("global_load_async_to_lds_b128 %0, %1, off"
               :
               : "v"(loff), "v"(ga)
               : "memory");
#else
  *(v8bf*)l = *(const v8bf*)g;
#endif
}

__device__ __forceinline__ void wait_async0() {
#if ASYNC_MODE == 2 && __has_builtin(__builtin_amdgcn_s_wait_asynccnt)
  __builtin_amdgcn_s_wait_asynccnt(0);
#elif ASYNC_MODE >= 1
  asm volatile("s_wait_asynccnt 0" ::: "memory");
#endif
}

// ---------------------------------------------------------------------------
// f32 -> bf16 elementwise convert
// ---------------------------------------------------------------------------
__global__ __launch_bounds__(256) void cvt_f32_bf16(const float* __restrict__ in,
                                                    __bf16* __restrict__ out,
                                                    int n) {
  int i = blockIdx.x * blockDim.x + threadIdx.x;
  int stride = gridDim.x * blockDim.x;
  for (; i < n; i += stride) out[i] = (__bf16)in[i];
}

// f32 [K][N] -> bf16 transposed [N][K] (one-time; lets GEMM stage W tiles as
// contiguous 16B chunks, required for async b128 copies)
__global__ __launch_bounds__(256) void cvt_f32_bf16_T(
    const float* __restrict__ in, __bf16* __restrict__ out, int K, int N) {
  int i = blockIdx.x * blockDim.x + threadIdx.x;
  int stride = gridDim.x * blockDim.x;
  int total = K * N;
  for (; i < total; i += stride) {
    int k = i / N, n = i - k * N;
    out[(size_t)n * K + k] = (__bf16)in[i];
  }
}

// ---------------------------------------------------------------------------
// GEMM: C[M,N] = act(A[M,K] * W[K,N] + bias[N]) with W given transposed
// (Wt[N][K], bf16). Block: 256 threads = 8 waves (2 M x 4 N). Block tile
// 128x128, wave tile 64x32, K-step 32. Double-buffered LDS with async
// memory->LDS copies overlapped with WMMA compute; one barrier per k-step.
// Grid: (N/128, M/128).
// ---------------------------------------------------------------------------
template <bool RELU, bool OUT_BF16>
__global__ __launch_bounds__(256) void gemm_bf16_kernel(
    const __bf16* __restrict__ A, const __bf16* __restrict__ Wt,
    const float* __restrict__ bias, float* __restrict__ Cf,
    __bf16* __restrict__ Cb, int M, int N, int K) {
  __shared__ __align__(16) __bf16 ldsA[2][128 * 32];  // [m][k]
  __shared__ __align__(16) __bf16 ldsW[2][128 * 32];  // [n][k]

  const int tid  = threadIdx.x;
  const int lane = tid & 31;
  const int w    = tid >> 5;
  const int wm   = w & 1;
  const int wn   = w >> 1;
  const int half = lane >> 4;
  const int nl   = lane & 15;

  const int m0 = blockIdx.y * 128;
  const int n0 = blockIdx.x * 128;

  v8f acc[4][2];
#pragma unroll
  for (int ms = 0; ms < 4; ++ms)
#pragma unroll
    for (int ns = 0; ns < 2; ++ns) acc[ms][ns] = vzero8();

  // Stage one 128x32 A tile + 128x32 Wt tile: 512 chunks of 16B each,
  // 2 chunks per thread per tile.
  auto stage = [&](int buf, int k0) {
#pragma unroll
    for (int i = 0; i < 2; ++i) {
      int c  = tid + 256 * i;         // 0..511
      int r  = c >> 2;                // 0..127
      int c8 = (c & 3) * 8;           // 0,8,16,24
      copy16_async(&A[(size_t)(m0 + r) * K + k0 + c8], &ldsA[buf][r * 32 + c8]);
      copy16_async(&Wt[(size_t)(n0 + r) * K + k0 + c8], &ldsW[buf][r * 32 + c8]);
    }
  };

  const int nk = K / 32;
  stage(0, 0);
  wait_async0();
  __syncthreads();

  for (int t = 0; t < nk; ++t) {
    const int cur = t & 1;
    if (t + 1 < nk) stage(cur ^ 1, (t + 1) * 32);
    if (t + 2 < nk) {
      __builtin_prefetch(&A[(size_t)(m0 + (tid >> 1)) * K + (t + 2) * 32], 0, 0);
      __builtin_prefetch(&Wt[(size_t)(n0 + (tid >> 1)) * K + (t + 2) * 32], 0, 0);
    }

    v16bf af[4], bfr[2];
#pragma unroll
    for (int ms = 0; ms < 4; ++ms) {
      int m = wm * 64 + ms * 16 + nl;
      af[ms] = ld8x2bf(&ldsA[cur][m * 32 + half * 8],
                       &ldsA[cur][m * 32 + 16 + half * 8]);
    }
#pragma unroll
    for (int ns = 0; ns < 2; ++ns) {
      int n = wn * 32 + ns * 16 + nl;
      bfr[ns] = ld8x2bf(&ldsW[cur][n * 32 + half * 8],
                        &ldsW[cur][n * 32 + 16 + half * 8]);
    }
#pragma unroll
    for (int ms = 0; ms < 4; ++ms)
#pragma unroll
      for (int ns = 0; ns < 2; ++ns)
        acc[ms][ns] = wmma_bf16(af[ms], bfr[ns], acc[ms][ns]);

    wait_async0();   // this wave's copies into the next buffer have landed
    __syncthreads(); // everyone done reading cur + next buffer fully staged
  }

  // Epilogue: C layout -> row = r + 8*half, col = nl
  float bv[2];
#pragma unroll
  for (int ns = 0; ns < 2; ++ns) bv[ns] = bias[n0 + wn * 32 + ns * 16 + nl];

#pragma unroll
  for (int ms = 0; ms < 4; ++ms) {
#pragma unroll
    for (int ns = 0; ns < 2; ++ns) {
      int col = n0 + wn * 32 + ns * 16 + nl;
#pragma unroll
      for (int r = 0; r < 8; ++r) {
        int row = m0 + wm * 64 + ms * 16 + r + 8 * half;
        float v = acc[ms][ns][r] + bv[ns];
        if (RELU) v = fmaxf(v, 0.0f);
        if (OUT_BF16)
          Cb[(size_t)row * N + col] = (__bf16)v;
        else
          Cf[(size_t)row * N + col] = v;
      }
    }
  }
}

// ---------------------------------------------------------------------------
// Flash-attention forward: O = softmax(Q K^T / sqrt(HD)) V, per (b,h).
// Q,K,V,O: bf16 in [B,S,D] layout. Block: 128 threads = 4 waves, each wave
// owns one 16-row q tile. kv tiles of 32 with online softmax. K tile staged
// via async copies; V tile transposed manually (scatter).
// Grid: (S/64, B*H).
// ---------------------------------------------------------------------------
__global__ __launch_bounds__(128) void attn_kernel(
    const __bf16* __restrict__ Q, const __bf16* __restrict__ K,
    const __bf16* __restrict__ V, __bf16* __restrict__ O) {
  __shared__ __align__(16) __bf16 ldsK[32 * 64];     // [kv][d]
  __shared__ __align__(16) __bf16 ldsVt[64 * 32];    // [d][kv]
  __shared__ __align__(16) __bf16 ldsP[4][16 * 32];  // per-wave P transpose

  const int bh   = blockIdx.y;
  const int b    = bh >> 4;
  const int h    = bh & 15;
  const int q0   = blockIdx.x * 64;
  const int tid  = threadIdx.x;
  const int lane = tid & 31;
  const int wv   = tid >> 5;
  const int half = lane >> 4;
  const int nl   = lane & 15;

  const int qrow = q0 + wv * 16 + nl;
  const __bf16* qp = Q + (size_t)(b * S_ + qrow) * D_ + h * HD_;
  v16bf aq[2];
#pragma unroll
  for (int kc = 0; kc < 2; ++kc)
    aq[kc] = ld8x2bf(qp + kc * 32 + half * 8, qp + kc * 32 + 16 + half * 8);

  float mx[8], ls[8];
  v8f acc[4];
#pragma unroll
  for (int r = 0; r < 8; ++r) { mx[r] = -3.0e38f; ls[r] = 0.0f; }
#pragma unroll
  for (int n = 0; n < 4; ++n) acc[n] = vzero8();

  const float scale = 0.125f;  // 1/sqrt(64)

  for (int kv0 = 0; kv0 < S_; kv0 += 32) {
    __syncthreads();
    // K tile (row-major, contiguous): 256 chunks of 16B, 2 per thread, async.
#pragma unroll
    for (int i = 0; i < 2; ++i) {
      int c  = tid + 128 * i;          // 0..255
      int r  = c >> 3;                 // 0..31
      int c8 = (c & 7) * 8;            // 0..56
      copy16_async(&K[(size_t)(b * S_ + kv0 + r) * D_ + h * HD_ + c8],
                   &ldsK[r * 64 + c8]);
    }
    // V tile transposed (scatter): manual.
#pragma unroll
    for (int i = 0; i < 16; ++i) {
      int e = tid + 128 * i;
      int r = e >> 6, c = e & 63;
      ldsVt[c * 32 + r] = V[(size_t)(b * S_ + kv0 + r) * D_ + h * HD_ + c];
    }
    wait_async0();
    __syncthreads();

    // Scores S = Q * K^T : 16 x 32 (two 16-col tiles), K-dim = HD = 64.
    v8f s[2];
#pragma unroll
    for (int nt = 0; nt < 2; ++nt) {
      v8f sc = vzero8();
#pragma unroll
      for (int kc = 0; kc < 2; ++kc) {
        v16bf bk = ld8x2bf(&ldsK[(nt * 16 + nl) * 64 + kc * 32 + half * 8],
                           &ldsK[(nt * 16 + nl) * 64 + kc * 32 + 16 + half * 8]);
        sc = wmma_bf16(aq[kc], bk, sc);
      }
      s[nt] = sc;
    }

    // Online softmax per row (row = r + 8*half, spread over 16 lanes of half)
#pragma unroll
    for (int r = 0; r < 8; ++r) {
      float a0 = s[0][r] * scale;
      float a1 = s[1][r] * scale;
      float mt = fmaxf(a0, a1);
#pragma unroll
      for (int msk = 1; msk < 16; msk <<= 1)
        mt = fmaxf(mt, __shfl_xor(mt, msk, 32));
      float mnew = fmaxf(mx[r], mt);
      float p0 = __expf(a0 - mnew);
      float p1 = __expf(a1 - mnew);
      float corr = __expf(mx[r] - mnew);
      mx[r] = mnew;
      float rs = p0 + p1;
#pragma unroll
      for (int msk = 1; msk < 16; msk <<= 1) rs += __shfl_xor(rs, msk, 32);
      ls[r] = ls[r] * corr + rs;
#pragma unroll
      for (int n = 0; n < 4; ++n) acc[n][r] *= corr;
      int m = r + 8 * half;
      ldsP[wv][m * 32 + nl]      = (__bf16)p0;
      ldsP[wv][m * 32 + 16 + nl] = (__bf16)p1;
    }
    __syncthreads();

    // O += P (16x32) * V (32x64)
    v16bf aP = ld8x2bf(&ldsP[wv][nl * 32 + half * 8],
                       &ldsP[wv][nl * 32 + 16 + half * 8]);
#pragma unroll
    for (int n = 0; n < 4; ++n) {
      v16bf bv = ld8x2bf(&ldsVt[(n * 16 + nl) * 32 + half * 8],
                         &ldsVt[(n * 16 + nl) * 32 + 16 + half * 8]);
      acc[n] = wmma_bf16(aP, bv, acc[n]);
    }
  }

  // Normalize and store (bf16, [B,S,D] layout)
#pragma unroll
  for (int n = 0; n < 4; ++n) {
#pragma unroll
    for (int r = 0; r < 8; ++r) {
      int m = r + 8 * half;
      float v = acc[n][r] / ls[r];
      O[(size_t)(b * S_ + q0 + wv * 16 + m) * D_ + h * HD_ + n * 16 + nl] =
          (__bf16)v;
    }
  }
}

// ---------------------------------------------------------------------------
// Residual add + LayerNorm over last dim (D=1024). One block per row.
// ---------------------------------------------------------------------------
template <bool WRITE_BF16>
__global__ __launch_bounds__(256) void add_ln_kernel(
    const float* __restrict__ x, const float* __restrict__ res,
    const float* __restrict__ g, const float* __restrict__ bta,
    float* __restrict__ outf, __bf16* __restrict__ outb) {
  __shared__ float rS[8], rSS[8];
  const int row  = blockIdx.x;
  const int tid  = threadIdx.x;
  const int lane = tid & 31;
  const int w    = tid >> 5;
  const float* xr = x + (size_t)row * D_;
  const float* rr = res + (size_t)row * D_;

  float vals[4];
  float s = 0.0f, ss = 0.0f;
#pragma unroll
  for (int i = 0; i < 4; ++i) {
    int c = tid + 256 * i;
    float v = xr[c] + rr[c];
    vals[i] = v;
    s += v;
    ss += v * v;
  }
#pragma unroll
  for (int msk = 16; msk >= 1; msk >>= 1) {
    s += __shfl_xor(s, msk, 32);
    ss += __shfl_xor(ss, msk, 32);
  }
  if (lane == 0) { rS[w] = s; rSS[w] = ss; }
  __syncthreads();
  if (tid == 0) {
    float S = 0.0f, SS = 0.0f;
#pragma unroll
    for (int i = 0; i < 8; ++i) { S += rS[i]; SS += rSS[i]; }
    rS[0] = S;
    rSS[0] = SS;
  }
  __syncthreads();
  const float mu  = rS[0] * (1.0f / D_);
  const float var = rSS[0] * (1.0f / D_) - mu * mu;
  const float inv = rsqrtf(var + 1e-5f);
#pragma unroll
  for (int i = 0; i < 4; ++i) {
    int c = tid + 256 * i;
    float v = (vals[i] - mu) * inv * g[c] + bta[c];
    outf[(size_t)row * D_ + c] = v;
    if (WRITE_BF16) outb[(size_t)row * D_ + c] = (__bf16)v;
  }
}

// ---------------------------------------------------------------------------
// Launcher
// ---------------------------------------------------------------------------
extern "C" void kernel_launch(void* const* d_in, const int* in_sizes, int n_in,
                              void* d_out, int out_size, void* d_ws,
                              size_t ws_size, hipStream_t stream) {
  (void)in_sizes; (void)n_in; (void)out_size; (void)ws_size;

  const float* src = (const float*)d_in[0];
  const float* Wq  = (const float*)d_in[1];
  const float* bq  = (const float*)d_in[2];
  const float* Wk  = (const float*)d_in[3];
  const float* bk  = (const float*)d_in[4];
  const float* Wv  = (const float*)d_in[5];
  const float* bv  = (const float*)d_in[6];
  const float* Wo  = (const float*)d_in[7];
  const float* bo  = (const float*)d_in[8];
  const float* g1  = (const float*)d_in[9];
  const float* b1  = (const float*)d_in[10];
  const float* W1  = (const float*)d_in[11];
  const float* bf1 = (const float*)d_in[12];
  const float* W2  = (const float*)d_in[13];
  const float* bf2 = (const float*)d_in[14];
  const float* g2  = (const float*)d_in[15];
  const float* b2  = (const float*)d_in[16];
  float* out = (float*)d_out;

  // Workspace carve-up (256B aligned)
  char* wsp = (char*)d_ws;
  size_t off = 0;
  auto alloc = [&](size_t bytes) -> void* {
    void* p = wsp + off;
    off = (off + bytes + 255) & ~(size_t)255;
    return p;
  };
  __bf16* srcb  = (__bf16*)alloc((size_t)M_ * D_ * 2);
  __bf16* wqt   = (__bf16*)alloc((size_t)D_ * D_ * 2);   // [N][K] transposed
  __bf16* wkt   = (__bf16*)alloc((size_t)D_ * D_ * 2);
  __bf16* wvt   = (__bf16*)alloc((size_t)D_ * D_ * 2);
  __bf16* wot   = (__bf16*)alloc((size_t)D_ * D_ * 2);
  __bf16* w1t   = (__bf16*)alloc((size_t)D_ * PF_ * 2);  // [PF][D]
  __bf16* w2t   = (__bf16*)alloc((size_t)PF_ * D_ * 2);  // [D][PF]
  __bf16* qb    = (__bf16*)alloc((size_t)M_ * D_ * 2);
  __bf16* kb    = (__bf16*)alloc((size_t)M_ * D_ * 2);
  __bf16* vb    = (__bf16*)alloc((size_t)M_ * D_ * 2);
  __bf16* xb    = (__bf16*)alloc((size_t)M_ * D_ * 2);
  float*  projf = (float*)alloc((size_t)M_ * D_ * 4);    // reused for FFN2 out
  float*  src1f = (float*)alloc((size_t)M_ * D_ * 4);
  __bf16* src1b = (__bf16*)alloc((size_t)M_ * D_ * 2);
  __bf16* ff1b  = (__bf16*)alloc((size_t)M_ * PF_ * 2);

  // 1) Convert inputs to bf16 (weights transposed to [N][K])
  cvt_f32_bf16<<<2048, 256, 0, stream>>>(src, srcb, M_ * D_);
  cvt_f32_bf16_T<<<1024, 256, 0, stream>>>(Wq, wqt, D_, D_);
  cvt_f32_bf16_T<<<1024, 256, 0, stream>>>(Wk, wkt, D_, D_);
  cvt_f32_bf16_T<<<1024, 256, 0, stream>>>(Wv, wvt, D_, D_);
  cvt_f32_bf16_T<<<1024, 256, 0, stream>>>(Wo, wot, D_, D_);
  cvt_f32_bf16_T<<<2048, 256, 0, stream>>>(W1, w1t, D_, PF_);
  cvt_f32_bf16_T<<<2048, 256, 0, stream>>>(W2, w2t, PF_, D_);

  // 2) QKV projections (bf16 out, bias folded)
  dim3 gQKV(D_ / 128, M_ / 128);
  gemm_bf16_kernel<false, true><<<gQKV, 256, 0, stream>>>(
      srcb, wqt, bq, nullptr, qb, M_, D_, D_);
  gemm_bf16_kernel<false, true><<<gQKV, 256, 0, stream>>>(
      srcb, wkt, bk, nullptr, kb, M_, D_, D_);
  gemm_bf16_kernel<false, true><<<gQKV, 256, 0, stream>>>(
      srcb, wvt, bv, nullptr, vb, M_, D_, D_);

  // 3) Attention (flash, online softmax)
  attn_kernel<<<dim3(S_ / 64, B_ * H_), 128, 0, stream>>>(qb, kb, vb, xb);

  // 4) Output projection (f32 out), residual + LN1 (f32 + bf16 copies)
  gemm_bf16_kernel<false, false><<<gQKV, 256, 0, stream>>>(
      xb, wot, bo, projf, nullptr, M_, D_, D_);
  add_ln_kernel<true><<<M_, 256, 0, stream>>>(projf, src, g1, b1, src1f, src1b);

  // 5) FFN: ReLU(src1 @ W1 + bf1) @ W2 + bf2, residual + LN2 -> out
  gemm_bf16_kernel<true, true><<<dim3(PF_ / 128, M_ / 128), 256, 0, stream>>>(
      src1b, w1t, bf1, nullptr, ff1b, M_, PF_, D_);
  gemm_bf16_kernel<false, false><<<gQKV, 256, 0, stream>>>(
      ff1b, w2t, bf2, projf, nullptr, M_, D_, PF_);
  add_ln_kernel<false><<<M_, 256, 0, stream>>>(projf, src1f, g2, b2, out,
                                               nullptr);
}
